// RaBitQ_62586263437631
// MI455X (gfx1250) — matrix operand order repaired and checked by reading, not compiled, not run on hardware
//
#include <hip/hip_runtime.h>

typedef _Float16 v16h __attribute__((ext_vector_type(16)));
typedef _Float16 v4h  __attribute__((ext_vector_type(4)));
typedef float    v8f  __attribute__((ext_vector_type(8)));
typedef float    v2f  __attribute__((ext_vector_type(2)));

#define DVEC 128
#define ROWS_TOTAL (4*16*4096)          /* 262144 vectors */
#define WAVES_PER_BLOCK 8
#define THREADS (WAVES_PER_BLOCK*32)

/* dynamic-LDS layout (bytes): exactly 64 KB */
#define P16_OFF   0                      /* 128x128 f16 row-major   (32 KB) */
#define PT16_OFF  (128*128*2)            /* 128x128 f16 transposed  (32 KB) */
#define SMEM_BYTES (2*128*128*2)

__global__ __launch_bounds__(THREADS)
void rabitq_fwd_kernel(const float* __restrict__ x, const float* __restrict__ P,
                       float* __restrict__ xhat, unsigned char* __restrict__ packed_out,
                       _Float16* __restrict__ norms_out, _Float16* __restrict__ x0_out)
{
    extern __shared__ char smem[];
    _Float16* P16  = (_Float16*)(smem + P16_OFF);
    _Float16* PT16 = (_Float16*)(smem + PT16_OFF);

    const int tid  = threadIdx.x;
    const int wave = tid >> 5;
    const int lane = tid & 31;

    /* ---- stage P: f32 -> f16, row-major + transposed (vectorized) ---- */
    for (int i = tid; i < (128*128)/4; i += THREADS) {
        float4 v = ((const float4*)P)[i];
        int base = i * 4;
        int r = base >> 7, c = base & 127;
        v4h h;
        h[0] = (_Float16)v.x; h[1] = (_Float16)v.y;
        h[2] = (_Float16)v.z; h[3] = (_Float16)v.w;
        *(v4h*)(P16 + base) = h;               /* one ds_store_b64 */
        PT16[(c+0)*128 + r] = h[0];
        PT16[(c+1)*128 + r] = h[1];
        PT16[(c+2)*128 + r] = h[2];
        PT16[(c+3)*128 + r] = h[3];
    }
    __syncthreads();

    const int  tile    = blockIdx.x * WAVES_PER_BLOCK + wave;
    const long rowbase = (long)tile * 16;

    const int m   = lane & 15;     /* matrix row this lane feeds (A layout) */
    const int mm  = m & 7;
    const int hi  = lane >> 4;     /* lane half: selects K sub-range        */
    const int hi8 = hi << 3;

    /* ---- load 16x128 x tile straight into WMMA A layout (RAW x, f16) ---- */
    const float* xr = x + (rowbase + m) * (long)DVEC;
    float4 f[16];
    #pragma unroll
    for (int kc = 0; kc < 4; ++kc) {
        int kb = kc*32 + hi*16;
        f[kc*4+0] = *(const float4*)(xr + kb);
        f[kc*4+1] = *(const float4*)(xr + kb + 4);
        f[kc*4+2] = *(const float4*)(xr + kb + 8);
        f[kc*4+3] = *(const float4*)(xr + kb + 12);
    }

    /* A = f16(x) directly: signs are scale-invariant; 1/norm folded into x0 */
    v16h a[4];
    #pragma unroll
    for (int kc = 0; kc < 4; ++kc) {
        #pragma unroll
        for (int q = 0; q < 4; ++q) {
            float4 v = f[kc*4+q];
            a[kc][q*4+0] = (_Float16)v.x;
            a[kc][q*4+1] = (_Float16)v.y;
            a[kc][q*4+2] = (_Float16)v.z;
            a[kc][q*4+3] = (_Float16)v.w;
        }
    }

    /* norms: packed f32 FMAs (v_pk_fma_f32) */
    v2f ssv = {0.f, 0.f};
    #pragma unroll
    for (int i = 0; i < 16; ++i) {
        v2f lo = { f[i].x, f[i].y }, hiw = { f[i].z, f[i].w };
        ssv += lo * lo;
        ssv += hiw * hiw;
    }
    float ss = ssv[0] + ssv[1];
    ss += __shfl_xor(ss, 16, 32);                 /* combine the two K-halves of row m */
    float norm = fmaxf(sqrtf(ss), 1e-8f);
    float rn   = 1.0f / norm;
    _Float16 nh = (_Float16)norm;                 /* norms16 for row m */

    /* ---- matmul 1: x_rot_raw = x @ P^T ; harvest signs + |x_rot_raw| ---- */
    float    part[8]   = {0,0,0,0,0,0,0,0};
    unsigned chunk[8];                       /* 16 sign bits of row m, per e-block */
    #pragma unroll
    for (int eb = 0; eb < 8; ++eb) {
        v8f acc = {0,0,0,0,0,0,0,0};
        const _Float16* brow = P16 + (eb*16 + m)*128 + hi*16;   /* B[k][n]=P[e][k] */
        #pragma unroll
        for (int kc = 0; kc < 4; ++kc) {
            v16h b = *(const v16h*)(brow + kc*32);
            acc = __builtin_amdgcn_wmma_f32_16x16x32_f16(
                    false, a[kc], false, b, (short)0, acc, false, false);
        }
        unsigned bal[8];
        #pragma unroll
        for (int r = 0; r < 8; ++r) {
            bal[r] = __builtin_amdgcn_ballot_w32(acc[r] >= 0.0f);
            part[r] += fabsf(acc[r]);
        }
        /* wave-uniform ballots: each lane selects the one for its own row m */
        unsigned sel = bal[0];
        #pragma unroll
        for (int r = 1; r < 8; ++r) sel = (mm == r) ? bal[r] : sel;
        chunk[eb] = (m >= 8) ? (sel >> 16) : (sel & 0xFFFFu);
    }

    /* 16 packed bytes of row m as 4 u32 words (little-endian byte order) */
    unsigned w0 = chunk[0] | (chunk[1] << 16);
    unsigned w1 = chunk[2] | (chunk[3] << 16);
    unsigned w2 = chunk[4] | (chunk[5] << 16);
    unsigned w3 = chunk[6] | (chunk[7] << 16);

    /* per-row scale factors for rows hi8+r, fetched once via lane permute */
    float normf = (float)nh;           /* fp16-rounded norm, as dequant uses */
    float rnr[8], nf[8];
    #pragma unroll
    for (int r = 0; r < 8; ++r) {
        rnr[r] = __shfl(rn,    hi8 + r, 32);
        nf[r]  = __shfl(normf, hi8 + r, 32);
    }

    /* ---- x0 = mean |x_rot_raw| / norm per row: in-half reduce + route ---- */
    float x0f[8];
    #pragma unroll
    for (int r = 0; r < 8; ++r) {
        float s = part[r];
        s += __shfl_xor(s, 1, 32);
        s += __shfl_xor(s, 2, 32);
        s += __shfl_xor(s, 4, 32);
        s += __shfl_xor(s, 8, 32);
        x0f[r] = s * (1.0f/128.0f) * rnr[r];   /* row hi8+r, uniform within half */
    }
    float x0own = x0f[0];
    #pragma unroll
    for (int r = 1; r < 8; ++r) x0own = (mm == r) ? x0f[r] : x0own;  /* row hi8+mm */
    float x0oth = __shfl_xor(x0own, 16, 32);                         /* other half's */
    float x0m   = ((m >> 3) == hi) ? x0own : x0oth;                  /* row m */
    _Float16 x0h = (_Float16)x0m;
    _Float16 x0n = -x0h;

    /* ---- side outputs: packed signs, norms f16, x0 f16 ---- */
    if (lane < 16) {
        long row = rowbase + lane;
        norms_out[row] = nh;
        x0_out[row]    = x0h;
        uint4 pk; pk.x = w0; pk.y = w1; pk.z = w2; pk.w = w3;
        *(uint4*)(packed_out + row*16) = pk;
    }

    /* ---- rebuild A2 = signs * x0 (f16) from the packed bits ---- */
    v16h a2[4];
    #pragma unroll
    for (int kc = 0; kc < 4; ++kc) {
        unsigned w  = (kc == 0) ? w0 : (kc == 1) ? w1 : (kc == 2) ? w2 : w3;
        unsigned b0 = (w >> (hi*16))     & 0xFFu;   /* byte kc*4 + 2*hi     */
        unsigned b1 = (w >> (hi*16 + 8)) & 0xFFu;   /* byte kc*4 + 2*hi + 1 */
        #pragma unroll
        for (int e = 0; e < 8; ++e) {
            a2[kc][e]   = ((b0 >> e) & 1u) ? x0h : x0n;
            a2[kc][e+8] = ((b1 >> e) & 1u) ? x0h : x0n;
        }
    }

    /* ---- matmul 2: x_hat = (signs*x0) @ P * norm16 ---- */
    float* orow = xhat + rowbase * (long)DVEC;
    #pragma unroll
    for (int eb = 0; eb < 8; ++eb) {
        v8f acc = {0,0,0,0,0,0,0,0};
        const _Float16* brow = PT16 + (eb*16 + m)*128 + hi*16;  /* B[k][n]=P[k][e] */
        #pragma unroll
        for (int kc = 0; kc < 4; ++kc) {
            v16h b = *(const v16h*)(brow + kc*32);
            acc = __builtin_amdgcn_wmma_f32_16x16x32_f16(
                    false, a2[kc], false, b, (short)0, acc, false, false);
        }
        int col = eb*16 + m;
        #pragma unroll
        for (int r = 0; r < 8; ++r)
            orow[(long)(hi8 + r)*DVEC + col] = acc[r] * nf[r];
    }
}

extern "C" void kernel_launch(void* const* d_in, const int* in_sizes, int n_in,
                              void* d_out, int out_size, void* d_ws, size_t ws_size,
                              hipStream_t stream) {
    const float* x = (const float*)d_in[0];   /* (4,16,4096,128) f32 */
    const float* P = (const float*)d_in[1];   /* (128,128) f32       */

    char* out = (char*)d_out;
    float*         xhat   = (float*)out;                                          /* 134,217,728 B */
    unsigned char* packed = (unsigned char*)(out + (size_t)ROWS_TOTAL*DVEC*4);    /*   4,194,304 B */
    _Float16*      norms  = (_Float16*)(out + (size_t)ROWS_TOTAL*DVEC*4
                                            + (size_t)ROWS_TOTAL*16);
    _Float16*      x0     = norms + ROWS_TOTAL;

    const int tiles  = ROWS_TOTAL / 16;              /* 16384 */
    const int blocks = tiles / WAVES_PER_BLOCK;      /* 2048  */

    rabitq_fwd_kernel<<<blocks, THREADS, SMEM_BYTES, stream>>>(x, P, xhat, packed, norms, x0);
}
